// Corr_block_84086869721617
// MI455X (gfx1250) — compile-verified, matrix operand order checked by reading, not run on hardware
//
#include <hip/hip_runtime.h>

typedef __attribute__((ext_vector_type(16))) _Float16 v16h;
typedef __attribute__((ext_vector_type(8)))  _Float16 v8h;
typedef __attribute__((ext_vector_type(8)))  float    v8f;

#define BATCH 4
#define CH    64
#define HW    128
#define NXY   26
#define PN    (NXY*NXY)   // 676 query positions per batch
#define RWIN  17          // search window side
#define RAREA (RWIN*RWIN) // 289
#define CPAD  72          // padded channel pitch (16B-aligned rows, spreads banks)
#define LOUT  15          // correlation output side
#define MPAD  225         // valid output positions
#define KTOT  576         // 3*3*64 contraction depth
#define OUP   16          // upsampled output side
#define EPSC  1.0132789611816406e-06f  // fp16(1e-6)

#define SHUF16(lo, hi) __builtin_shufflevector(lo, hi, 0,1,2,3,4,5,6,7,8,9,10,11,12,13,14,15)

__global__ void __launch_bounds__(256)
corr_wmma_kernel(const float* __restrict__ f1,
                 const float* __restrict__ f2,
                 float* __restrict__ out)
{
    __shared__ alignas(16) _Float16 roi[RAREA * CPAD]; // [u][v][c], fp16 window
    __shared__ alignas(16) _Float16 qv[KTOT];          // normalized q, k=(i*3+j)*64+c
    __shared__ float roi2[RAREA];
    __shared__ float numb[256];                        // 225 read back
    __shared__ float cv[LOUT * LOUT];
    __shared__ float ssum;

    const int tid = threadIdx.x;
    const int bp  = blockIdx.x;
    const int b   = bp / PN;
    const int p   = bp - b * PN;
    const int px  = p / NXY;
    const int py  = p - px * NXY;

    const float* f1b = f1 + (size_t)b * CH * HW * HW;
    const float* f2b = f2 + (size_t)b * CH * HW * HW;

    if (tid == 0) ssum = 0.f;

    const int h0 = px * 5 - 8, w0 = py * 5 - 8;

    // Warm L2/WGP$ for this block's window (global_prefetch_b8).
    {
        int c  = tid & 63;
        int hc = h0 < 0 ? 0 : h0;
        int wc = w0 < 0 ? 0 : w0;
        __builtin_prefetch(f2b + (size_t)c * HW * HW + hc * HW + wc, 0, 0);
    }

    // ---- stage roi window (17x17x64) -> LDS [u][v][c], zero-padded, fp16 ----
    for (int e = tid; e < CH * RAREA; e += 256) {
        int c   = e / RAREA;           // global-coalesced within 17-wide rows
        int rem = e - c * RAREA;
        int u   = rem / RWIN;
        int v   = rem - u * RWIN;
        int h = h0 + u, w = w0 + v;
        float val = 0.f;
        if (h >= 0 && h < HW && w >= 0 && w < HW)
            val = f2b[(size_t)c * HW * HW + h * HW + w];
        roi[(u * RWIN + v) * CPAD + c] = (_Float16)val;
    }

    // ---- stage query (3x3x64), accumulate sum of squares ----
    float part = 0.f;
    const int qh0 = px * 5 - 1, qw0 = py * 5 - 1;
    for (int e = tid; e < KTOT; e += 256) {
        int ij = e >> 6;
        int c  = e & 63;
        int i  = ij / 3, j = ij - 3 * i;
        int h = qh0 + i, w = qw0 + j;
        float val = 0.f;
        if (h >= 0 && h < HW && w >= 0 && w < HW)
            val = f1b[(size_t)c * HW * HW + h * HW + w];
        _Float16 hv = (_Float16)val;
        qv[e] = hv;
        float fv = (float)hv;
        part += fv * fv;
    }
    __syncthreads();
    atomicAdd(&ssum, part);            // ds_add_f32 reduction
    __syncthreads();
    const float qinv = 1.f / (sqrtf(ssum) + EPSC);
    for (int e = tid; e < KTOT; e += 256)
        qv[e] = (_Float16)((float)qv[e] * qinv);

    // ---- per-(u,v) window energy, vectorized over channels ----
    for (int uv = tid; uv < RAREA; uv += 256) {
        float acc = 0.f;
#pragma unroll
        for (int c8 = 0; c8 < 8; ++c8) {
            v8h r = *(const v8h*)&roi[uv * CPAD + c8 * 8];
#pragma unroll
            for (int t = 0; t < 8; ++t) {
                float f = (float)r[t];
                acc += f * f;
            }
        }
        roi2[uv] = acc;
    }
    __syncthreads();

    // ---- implicit-GEMM via v_wmma_f32_16x16x32_f16 ----
    // M = output position (x*15+y) in 16 tiles of 16 (tile 15 = discard pad),
    // K = (i,j,c) = 576 in 18 steps of 32, N = 1 (q broadcast; D cols 1..15 unused).
    const int lane = tid & 31;
    const int wave = tid >> 5;
    const int mloc = lane & 15;       // A row within tile
    const int g    = lane >> 4;       // 16-bit operand column-group
    const int m0 = wave * 32 + mloc;  // tile 2*wave
    const int m1 = m0 + 16;           // tile 2*wave+1
    // Rows m >= 225 produce garbage that is never read back: just clamp.
    const int mc0 = (m0 < MPAD) ? m0 : (MPAD - 1);
    const int mc1 = (m1 < MPAD) ? m1 : (MPAD - 1);
    const int x0 = mc0 / 15, y0 = mc0 - 15 * x0;
    const int x1 = mc1 / 15, y1 = mc1 - 15 * x1;
    const int gco = g * 8;            // +8g channel offset per slot map

    v8f acc0 = {};
    v8f acc1 = {};
#pragma unroll
    for (int i = 0; i < 3; ++i) {
#pragma unroll
        for (int j = 0; j < 3; ++j) {
            const int ij  = i * 3 + j;
            const int ar0 = ((x0 + i) * RWIN + (y0 + j)) * CPAD + gco;
            const int ar1 = ((x1 + i) * RWIN + (y1 + j)) * CPAD + gco;
            const int br  = ij * 64 + gco;
#pragma unroll
            for (int cb = 0; cb < 2; ++cb) {
                const int co = cb * 32;
                // slot map: VGPR0-3 -> K 0-7 (+8g), VGPR4-7 -> K 16-23 (+8g):
                // two contiguous 16B runs -> two ds_load_b128 per fragment.
                v8h a0l = *(const v8h*)&roi[ar0 + co];
                v8h a0h = *(const v8h*)&roi[ar0 + co + 16];
                v8h a1l = *(const v8h*)&roi[ar1 + co];
                v8h a1h = *(const v8h*)&roi[ar1 + co + 16];
                v8h bl  = *(const v8h*)&qv[br + co];
                v8h bh  = *(const v8h*)&qv[br + co + 16];
                v16h afr0 = SHUF16(a0l, a0h);
                v16h afr1 = SHUF16(a1l, a1h);
                v16h bfr  = SHUF16(bl, bh);
                acc0 = __builtin_amdgcn_wmma_f32_16x16x32_f16(
                           false, afr0, false, bfr, (short)0, acc0, false, false);
                acc1 = __builtin_amdgcn_wmma_f32_16x16x32_f16(
                           false, afr1, false, bfr, (short)0, acc1, false, false);
            }
        }
    }

    // D column 0: lane 0 rows 0-7, lane 16 rows 8-15
    if (mloc == 0) {
        const int roff = g * 8;
#pragma unroll
        for (int r = 0; r < 8; ++r) {
            numb[wave * 32 + roff + r]      = acc0[r];
            numb[wave * 32 + 16 + roff + r] = acc1[r];
        }
    }
    __syncthreads();

    // ---- normalize by window energy, transpose (swapaxes(-1,-2)) ----
    for (int m = tid; m < MPAD; m += 256) {
        const int x = m / 15, y = m - 15 * x;
        float s2 = 0.f;
#pragma unroll
        for (int i = 0; i < 3; ++i)
#pragma unroll
            for (int j = 0; j < 3; ++j)
                s2 += roi2[(x + i) * RWIN + (y + j)];
        cv[y * LOUT + x] = numb[m] / (sqrtf(s2) + EPSC);
    }
    __syncthreads();

    // ---- align-corners bilinear 15 -> 16 ----
    {
        const int oy = tid >> 4, ox = tid & 15;
        const float sc  = 14.f / 15.f;
        const float pyf = oy * sc, pxf = ox * sc;
        const int yl = (int)pyf;
        const int xl = (int)pxf;
        const int yh = (yl + 1 > LOUT - 1) ? (LOUT - 1) : yl + 1;
        const int xh = (xl + 1 > LOUT - 1) ? (LOUT - 1) : xl + 1;
        const float wy = pyf - (float)yl, wx = pxf - (float)xl;
        const float v00 = cv[yl * LOUT + xl];
        const float v01 = cv[yl * LOUT + xh];
        const float v10 = cv[yh * LOUT + xl];
        const float v11 = cv[yh * LOUT + xh];
        const float left  = v00 * (1.f - wy) + v10 * wy;
        const float right = v01 * (1.f - wy) + v11 * wy;
        out[((size_t)bp * OUP + oy) * OUP + ox] = left * (1.f - wx) + right * wx;
    }
}

extern "C" void kernel_launch(void* const* d_in, const int* in_sizes, int n_in,
                              void* d_out, int out_size, void* d_ws, size_t ws_size,
                              hipStream_t stream)
{
    (void)in_sizes; (void)n_in; (void)d_ws; (void)ws_size; (void)out_size;
    const float* f1 = (const float*)d_in[0];
    const float* f2 = (const float*)d_in[1];
    float* out = (float*)d_out;
    corr_wmma_kernel<<<dim3(BATCH * PN), dim3(256), 0, stream>>>(f1, f2, out);
}
